// SDARMoeSparseMoeBlock_79474074845732
// MI455X (gfx1250) — matrix operand order repaired and checked by reading, not compiled, not run on hardware
//
#include <hip/hip_runtime.h>
#include <hip/hip_bf16.h>
#include <cstdint>
#include <cstddef>

#define S_TOK 8192
#define HDIM  2048
#define IDIM  2048
#define NEXP  8

typedef __attribute__((ext_vector_type(8)))  float   v8f;
typedef __attribute__((ext_vector_type(8)))  __bf16  v8bf;
typedef __attribute__((ext_vector_type(16))) __bf16  v16bf;

#define LDS_STR 72   // bf16 elements per LDS row: 64 data + 8 pad (144B, 16B aligned)

// ---------- helpers ----------
__device__ __forceinline__ unsigned pack_bf16x2(float a, float b) {
    unsigned ua = __float_as_uint(a), ub = __float_as_uint(b);
    ua = (ua + 0x7FFFu + ((ua >> 16) & 1u)) >> 16;   // RNE
    ub = (ub + 0x7FFFu + ((ub >> 16) & 1u)) >> 16;
    return (ua & 0xFFFFu) | (ub << 16);
}
__device__ __forceinline__ unsigned short f32_to_bf16(float a) {
    unsigned ua = __float_as_uint(a);
    ua = (ua + 0x7FFFu + ((ua >> 16) & 1u)) >> 16;
    return (unsigned short)ua;
}

// stage 16 consecutive fp32 -> 16 bf16 into LDS (two 16B stores)
__device__ __forceinline__ void stage16(const float* src, __bf16* dst) {
    const float4* s = (const float4*)src;
    float4 f0 = s[0], f1 = s[1], f2 = s[2], f3 = s[3];
    uint4 p0, p1;
    p0.x = pack_bf16x2(f0.x, f0.y); p0.y = pack_bf16x2(f0.z, f0.w);
    p0.z = pack_bf16x2(f1.x, f1.y); p0.w = pack_bf16x2(f1.z, f1.w);
    p1.x = pack_bf16x2(f2.x, f2.y); p1.y = pack_bf16x2(f2.z, f2.w);
    p1.z = pack_bf16x2(f3.x, f3.y); p1.w = pack_bf16x2(f3.z, f3.w);
    *(uint4*)dst       = p0;
    *(uint4*)(dst + 8) = p1;
}

// gfx1250 async memory -> LDS copy, tracked with ASYNCcnt (ISA 15.18.3 op 98).
// lds_off = low 32 bits of the generic shared pointer (LDS aperture: addr[31:0]).
__device__ __forceinline__ void async_copy_b128(unsigned lds_off, const void* gptr) {
    asm volatile("global_load_async_to_lds_b128 %0, %1, off"
                 :: "v"(lds_off), "v"((unsigned long long)(uintptr_t)gptr)
                 : "memory");
}
__device__ __forceinline__ void async_wait0() {
    asm volatile("s_wait_asynccnt 0x0" ::: "memory");
}

// A-matrix 16x32 bf16 fragment (ISA 7.12.2): lanes 0-15 hold {K0-7,K16-23},
// lanes 16-31 hold {K8-15,K24-31}; row = M index, ks = K/32 sub-slab.
__device__ __forceinline__ v16bf ld_fragA(const __bf16* tile, int row, int ks, int lane) {
    const __bf16* p = tile + row * LDS_STR + ks * 32 + ((lane >> 4) << 3);
    v8bf lo = *(const v8bf*)p;
    v8bf hi = *(const v8bf*)(p + 16);
    return __builtin_shufflevector(lo, hi, 0,1,2,3,4,5,6,7,8,9,10,11,12,13,14,15);
}
// B-matrix 32x16 bf16 fragment: lanes 0-15 hold {K0-7,K16-23} of column N=row,
// lanes 16-31 hold {K8-15,K24-31} (same K order as A).
__device__ __forceinline__ v16bf ld_fragB(const __bf16* tile, int row, int ks, int lane) {
    const __bf16* p = tile + row * LDS_STR + ks * 32 + ((lane >> 4) << 3);
    v8bf lo = *(const v8bf*)p;
    v8bf hi = *(const v8bf*)(p + 16);
    return __builtin_shufflevector(lo, hi, 0,1,2,3,4,5,6,7,8,9,10,11,12,13,14,15);
}

__device__ __forceinline__ v8f wmma_bf16(v16bf a, v16bf b, v8f c) {
    // (neg_a, A, neg_b, B, c_mod, C, reuse_a, reuse_b)
    return __builtin_amdgcn_wmma_f32_16x16x32_bf16(false, a, false, b, (short)0, c, false, false);
}

// ---------- zero-init ----------
__global__ void zero_f4(float4* p, int n4) {
    int i = blockIdx.x * blockDim.x + threadIdx.x;
    if (i < n4) p[i] = make_float4(0.f, 0.f, 0.f, 0.f);
}
__global__ void zero_i(int* p, int n) {
    int i = threadIdx.x;
    if (i < n) p[i] = 0;
}

// ---------- router: logits, softmax, top-2, slot assignment ----------
__global__ __launch_bounds__(256)
void router_kernel(const float* __restrict__ hs, const float* __restrict__ wg,
                   float* __restrict__ logits, int* __restrict__ counts,
                   int* __restrict__ perm, float* __restrict__ wslot) {
    __shared__ float sg[NEXP * HDIM];   // 64 KB: whole gate matrix
    const int tid = threadIdx.x, lane = tid & 31, wave = tid >> 5;
    for (int i = tid * 4; i < NEXP * HDIM; i += 256 * 4)
        *(float4*)&sg[i] = *(const float4*)&wg[i];
    __syncthreads();

    const int token = blockIdx.x * 8 + wave;
    const float* hrow = hs + (size_t)token * HDIM;
    float acc[NEXP];
#pragma unroll
    for (int e = 0; e < NEXP; ++e) acc[e] = 0.f;

    for (int k = lane * 4; k < HDIM; k += 128) {
        float4 h4 = *(const float4*)(hrow + k);
#pragma unroll
        for (int e = 0; e < NEXP; ++e) {
            float4 w4 = *(const float4*)&sg[e * HDIM + k];
            acc[e] += h4.x * w4.x + h4.y * w4.y + h4.z * w4.z + h4.w * w4.w;
        }
    }
#pragma unroll
    for (int e = 0; e < NEXP; ++e)
        for (int off = 16; off > 0; off >>= 1)
            acc[e] += __shfl_xor(acc[e], off, 32);

    if (lane == 0) {
        float m = acc[0];
#pragma unroll
        for (int e = 1; e < NEXP; ++e) m = fmaxf(m, acc[e]);
        float ex[NEXP];
#pragma unroll
        for (int e = 0; e < NEXP; ++e) ex[e] = __expf(acc[e] - m);
        int i0 = 0; float p0 = ex[0];
#pragma unroll
        for (int e = 1; e < NEXP; ++e) if (ex[e] > p0) { p0 = ex[e]; i0 = e; }
        int i1 = -1; float p1 = -1.f;
#pragma unroll
        for (int e = 0; e < NEXP; ++e) if (e != i0 && ex[e] > p1) { p1 = ex[e]; i1 = e; }
        float inv = 1.f / (p0 + p1);
#pragma unroll
        for (int e = 0; e < NEXP; ++e) logits[(size_t)token * NEXP + e] = acc[e];
        int s0 = atomicAdd(&counts[i0], 1);
        perm[i0 * S_TOK + s0] = token; wslot[i0 * S_TOK + s0] = p0 * inv;
        int s1 = atomicAdd(&counts[i1], 1);
        perm[i1 * S_TOK + s1] = token; wslot[i1 * S_TOK + s1] = p1 * inv;
    }
}

__global__ void scan_kernel(const int* __restrict__ counts, int* __restrict__ base) {
    if (threadIdx.x == 0) {
        int a = 0;
        for (int e = 0; e < NEXP; ++e) { base[e] = a; a += counts[e]; }
        base[NEXP] = a;
    }
}

// ---------- expert up-proj: act = silu(h@Wg^T) * (h@Wu^T) * combine_w ----------
__global__ __launch_bounds__(256)
void moe_up_gemm(const float* __restrict__ hs, const float* __restrict__ w1,
                 const int* __restrict__ counts, const int* __restrict__ base,
                 const int* __restrict__ perm, const float* __restrict__ wslot,
                 unsigned short* __restrict__ act) {
    const int e = blockIdx.z;
    const int rows_e = counts[e];
    const int m0 = blockIdx.y * 128;
    if (m0 >= rows_e) return;
    const int n0 = blockIdx.x * 64;
    const int base_e = base[e];
    const int tid = threadIdx.x, lane = tid & 31, wave = tid >> 5;
    const int wm = wave >> 1, wn = wave & 1;

    __shared__ __bf16 ldsA[128 * LDS_STR];   // token tile (128 x 64)
    __shared__ __bf16 ldsB[128 * LDS_STR];   // gate rows 0-63, up rows 64-127
    __shared__ float  wts[128];

    const int srow = tid >> 1;
    const int h0 = (tid & 1) * 2;            // this thread stages halves h0, h0+1
    int gr = m0 + srow; if (gr >= rows_e) gr = rows_e - 1;   // clamp: garbage rows masked later
    const float* asrc = hs + (size_t)perm[e * S_TOK + gr] * HDIM;
    const int frow = (srow < 64) ? (n0 + srow) : (IDIM + n0 + srow - 64);
    const float* bsrc = w1 + ((size_t)e * 2 * IDIM + frow) * HDIM;

    if (tid < 128) wts[tid] = (m0 + tid < rows_e) ? wslot[e * S_TOK + m0 + tid] : 0.f;

    v8f accG[2][2] = {};
    v8f accU[2][2] = {};

    for (int kk = 0; kk < HDIM; kk += 64) {
        __syncthreads();
        stage16(asrc + kk + h0 * 16,      &ldsA[srow * LDS_STR + h0 * 16]);
        stage16(asrc + kk + h0 * 16 + 16, &ldsA[srow * LDS_STR + h0 * 16 + 16]);
        stage16(bsrc + kk + h0 * 16,      &ldsB[srow * LDS_STR + h0 * 16]);
        stage16(bsrc + kk + h0 * 16 + 16, &ldsB[srow * LDS_STR + h0 * 16 + 16]);
        if (kk + 64 < HDIM) __builtin_prefetch(bsrc + kk + 64, 0, 1);  // global_prefetch
        __syncthreads();

#pragma unroll
        for (int ks = 0; ks < 2; ++ks) {
            v16bf a0 = ld_fragA(ldsA, wm * 32 +      (lane & 15), ks, lane);
            v16bf a1 = ld_fragA(ldsA, wm * 32 + 16 + (lane & 15), ks, lane);
            v16bf g0 = ld_fragB(ldsB, wn * 32 +      (lane & 15), ks, lane);
            v16bf g1 = ld_fragB(ldsB, wn * 32 + 16 + (lane & 15), ks, lane);
            v16bf u0 = ld_fragB(ldsB, 64 + wn * 32 +      (lane & 15), ks, lane);
            v16bf u1 = ld_fragB(ldsB, 64 + wn * 32 + 16 + (lane & 15), ks, lane);

            accG[0][0] = wmma_bf16(a0, g0, accG[0][0]);
            accG[0][1] = wmma_bf16(a0, g1, accG[0][1]);
            accG[1][0] = wmma_bf16(a1, g0, accG[1][0]);
            accG[1][1] = wmma_bf16(a1, g1, accG[1][1]);
            accU[0][0] = wmma_bf16(a0, u0, accU[0][0]);
            accU[0][1] = wmma_bf16(a0, u1, accU[0][1]);
            accU[1][0] = wmma_bf16(a1, u0, accU[1][0]);
            accU[1][1] = wmma_bf16(a1, u1, accU[1][1]);
        }
    }

    const int jb = n0 + wn * 32;
#pragma unroll
    for (int mi = 0; mi < 2; ++mi)
#pragma unroll
    for (int ni = 0; ni < 2; ++ni) {
        const int rb = wm * 32 + mi * 16 + ((lane >> 4) << 3);
        const int j  = jb + ni * 16 + (lane & 15);
#pragma unroll
        for (int v = 0; v < 8; ++v) {
            const int r = rb + v, grr = m0 + r;
            if (grr < rows_e) {
                float g = accG[mi][ni][v], u = accU[mi][ni][v];
                float a = (g / (1.f + __expf(-g))) * u * wts[r];
                act[(size_t)(base_e + grr) * IDIM + j] = f32_to_bf16(a);
            }
        }
    }
}

// ---------- expert down-proj + weighted scatter into out ----------
__global__ __launch_bounds__(256)
void moe_down_gemm(const unsigned short* __restrict__ act, const float* __restrict__ w2,
                   const int* __restrict__ counts, const int* __restrict__ base,
                   const int* __restrict__ perm, float* __restrict__ out) {
    const int e = blockIdx.z;
    const int rows_e = counts[e];
    const int m0 = blockIdx.y * 128;
    if (m0 >= rows_e) return;
    const int n0 = blockIdx.x * 64;
    const int base_e = base[e];
    const int tid = threadIdx.x, lane = tid & 31, wave = tid >> 5;
    const int wm = wave >> 1, wn = wave & 1;

    __shared__ __bf16 ldsA[128 * LDS_STR];
    __shared__ __bf16 ldsB[64 * LDS_STR];
    __shared__ int    toks[128];

    if (tid < 128) toks[tid] = (m0 + tid < rows_e) ? perm[e * S_TOK + m0 + tid] : -1;

    const int srow = tid >> 1;
    const int h0 = (tid & 1) * 2;
    int arow = m0 + srow; if (arow >= rows_e) arow = rows_e - 1;  // clamp
    const unsigned short* asrc = act + (size_t)(base_e + arow) * IDIM;
    const float* bsrc = w2 + ((size_t)e * HDIM + n0 + (srow & 63)) * IDIM;

    // LDS byte address of this thread's A staging slot (32 bf16 = 64B)
    const unsigned ldsa_off =
        (unsigned)(uintptr_t)&ldsA[srow * LDS_STR + h0 * 16];

    v8f acc[2][2] = {};

    for (int kk = 0; kk < IDIM; kk += 64) {
        __syncthreads();
        {   // A already bf16: async memory->LDS copy, no VGPR round trip
            const unsigned short* as = asrc + kk + h0 * 16;
            async_copy_b128(ldsa_off,      as);
            async_copy_b128(ldsa_off + 16, as + 8);
            async_copy_b128(ldsa_off + 32, as + 16);
            async_copy_b128(ldsa_off + 48, as + 24);
        }
        if (tid < 128) {
            stage16(bsrc + kk + h0 * 16,      &ldsB[srow * LDS_STR + h0 * 16]);
            stage16(bsrc + kk + h0 * 16 + 16, &ldsB[srow * LDS_STR + h0 * 16 + 16]);
        }
        if (kk + 64 < IDIM) __builtin_prefetch(bsrc + kk + 64, 0, 1);
        async_wait0();
        __syncthreads();

#pragma unroll
        for (int ks = 0; ks < 2; ++ks) {
            v16bf a0 = ld_fragA(ldsA, wm * 32 +      (lane & 15), ks, lane);
            v16bf a1 = ld_fragA(ldsA, wm * 32 + 16 + (lane & 15), ks, lane);
            v16bf b0 = ld_fragB(ldsB, wn * 32 +      (lane & 15), ks, lane);
            v16bf b1 = ld_fragB(ldsB, wn * 32 + 16 + (lane & 15), ks, lane);

            acc[0][0] = wmma_bf16(a0, b0, acc[0][0]);
            acc[0][1] = wmma_bf16(a0, b1, acc[0][1]);
            acc[1][0] = wmma_bf16(a1, b0, acc[1][0]);
            acc[1][1] = wmma_bf16(a1, b1, acc[1][1]);
        }
    }

#pragma unroll
    for (int mi = 0; mi < 2; ++mi)
#pragma unroll
    for (int ni = 0; ni < 2; ++ni) {
        const int rb = wm * 32 + mi * 16 + ((lane >> 4) << 3);
        const int n  = n0 + wn * 32 + ni * 16 + (lane & 15);
#pragma unroll
        for (int v = 0; v < 8; ++v) {
            const int t = toks[rb + v];
            if (t >= 0) unsafeAtomicAdd(&out[(size_t)t * HDIM + n], acc[mi][ni][v]);
        }
    }
}

// ---------- launch ----------
extern "C" void kernel_launch(void* const* d_in, const int* in_sizes, int n_in,
                              void* d_out, int out_size, void* d_ws, size_t ws_size,
                              hipStream_t stream) {
    const float* hs = (const float*)d_in[0];
    const float* wg = (const float*)d_in[1];
    const float* w1 = (const float*)d_in[2];
    const float* w2 = (const float*)d_in[3];
    // top_k (d_in[4]) is fixed at 2 per the reference constants

    float* out    = (float*)d_out;
    float* logits = out + (size_t)S_TOK * HDIM;

    char* ws = (char*)d_ws;
    int*   counts = (int*)ws;                                   // 16 ints
    int*   base   = (int*)(ws + 64);                            // 16 ints
    int*   perm   = (int*)(ws + 1024);                          // E*S ints   (256 KB)
    float* wslot  = (float*)(ws + 1024 + NEXP * S_TOK * 4);     // E*S floats (256 KB)
    unsigned short* act =
        (unsigned short*)(ws + 1024 + NEXP * S_TOK * 8);        // 2*S x I bf16 (64 MB)

    const int n4 = S_TOK * HDIM / 4;
    zero_f4<<<(n4 + 255) / 256, 256, 0, stream>>>((float4*)out, n4);
    zero_i <<<1, 64, 0, stream>>>(counts, 16);
    router_kernel<<<S_TOK / 8, 256, 0, stream>>>(hs, wg, logits, counts, perm, wslot);
    scan_kernel<<<1, 32, 0, stream>>>(counts, base);
    moe_up_gemm  <<<dim3(IDIM / 64, S_TOK / 128, NEXP), 256, 0, stream>>>(
        hs, w1, counts, base, perm, wslot, act);
    moe_down_gemm<<<dim3(HDIM / 64, S_TOK / 128, NEXP), 256, 0, stream>>>(
        act, w2, counts, base, perm, out);
}
